// Adapter_29515015258180
// MI455X (gfx1250) — matrix operand order repaired and checked by reading, not compiled, not run on hardware
//
#include <hip/hip_runtime.h>
#include <hip/hip_bf16.h>

#define SIZE 2048
#define HID  128
#define BATCH 4096
#define NTASK 8

typedef __attribute__((ext_vector_type(4)))  _Float16 v4h;
typedef __attribute__((ext_vector_type(8)))  _Float16 v8h;
typedef __attribute__((ext_vector_type(16))) _Float16 v16h;
typedef __attribute__((ext_vector_type(8)))  float    v8f;

#define CAT16(lo, hi) __builtin_shufflevector((lo), (hi), 0,1,2,3,4,5,6,7,8,9,10,11,12,13,14,15)

__device__ __forceinline__ int imin(int a, int b) { return a < b ? a : b; }

// ---------------- bucketing ----------------

__global__ void k_zero(int* ctrs) {
    int i = threadIdx.x;
    if (i < 16) ctrs[i] = 0;
}

__global__ void k_count(const int* __restrict__ task_id, int* __restrict__ counts) {
    int i = blockIdx.x * blockDim.x + threadIdx.x;
    if (i >= BATCH) return;
    int t = task_id[i];
    if (t >= 0 && t < NTASK) atomicAdd(&counts[t], 1);
}

__global__ void k_scan(const int* __restrict__ counts, int* __restrict__ offs) {
    if (threadIdx.x == 0) {
        int s = 0;
        for (int t = 0; t < NTASK; ++t) { offs[t] = s; s += counts[t]; }
    }
}

__global__ void k_scatter(const int* __restrict__ task_id, const int* __restrict__ offs,
                          int* __restrict__ cursors, int* __restrict__ perm) {
    int i = blockIdx.x * blockDim.x + threadIdx.x;
    if (i >= BATCH) return;
    int t = task_id[i];
    if (t >= 0 && t < NTASK) {
        int pos = offs[t] + atomicAdd(&cursors[t], 1);
        perm[pos] = i;
    }
}

// ---------------- residual copy: out = x ----------------

__global__ void k_copy(const float4* __restrict__ x, float4* __restrict__ out, int n4) {
    int i = blockIdx.x * blockDim.x + threadIdx.x;
    if (i < n4) out[i] = x[i];
}

// -------- pack weights f32 -> f16 in WMMA B-operand layout --------
// P[t][nTile][kt][lane][j] = W[t][kt*32 + (lane>>4)*16 + j][nTile*16 + (lane&15)]
__global__ void k_pack(const float* __restrict__ W, _Float16* __restrict__ P,
                       int Ktot, int Ntot) {
    int kTiles = Ktot >> 5;
    int nTiles = Ntot >> 4;
    long total = (long)NTASK * Ktot * Ntot;
    long tid = (long)blockIdx.x * blockDim.x + threadIdx.x;
    if (tid >= total) return;
    int nlo = (int)(tid & 15);
    int j   = (int)((tid >> 4) & 15);
    int hk  = (int)((tid >> 8) & 1);
    long blk = tid >> 9;
    int kt = (int)(blk % kTiles);
    long r2 = blk / kTiles;
    int nT = (int)(r2 % nTiles);
    int t  = (int)(r2 / nTiles);
    int lane = hk * 16 + nlo;
    int K = kt * 32 + hk * 16 + j;
    int n = nT * 16 + nlo;
    float v = W[((long)t * Ktot + K) * (long)Ntot + n];
    P[(blk * 32 + lane) * 16 + j] = (_Float16)v;
}

// ---------------- down projection + SiLU ----------------
// grid: (256 M-tiles, 8 tasks), block: 256 (8 waves; wave w owns nTile w)
#define RSX 1032  // LDS row stride in halves (1024 + 8 pad -> conflict-free b128)

__global__ __launch_bounds__(256) void k_down(
    const float* __restrict__ x, const _Float16* __restrict__ WdP,
    const float* __restrict__ bd, const int* __restrict__ perm,
    const int* __restrict__ counts, const int* __restrict__ offs,
    _Float16* __restrict__ Hp)
{
    __shared__ _Float16 xs[16 * RSX];   // 33 KB: 16 rows x 1024 K-halves (per phase)
    int t = blockIdx.y;
    int cnt = counts[t];
    int mBase = blockIdx.x * 16;
    if (mBase >= cnt) return;
    int sstart = offs[t];

    int tid = threadIdx.x;
    int lane = tid & 31;
    int wave = tid >> 5;           // nTile 0..7
    int l16 = lane & 15;
    int hk  = lane >> 4;

    float bias = bd[t * HID + wave * 16 + l16];
    v8f acc = { bias, bias, bias, bias, bias, bias, bias, bias };

    // staging role: row m, 16 threads per row
    int m  = tid >> 4;
    int c0 = (tid & 15) * 4;
    int grow = perm[sstart + imin(mBase + m, cnt - 1)];
    const float* xr = x + (long)grow * SIZE;

    for (int phase = 0; phase < 2; ++phase) {
        __syncthreads();
        for (int c = c0; c < 1024; c += 64) {
            float4 v = *(const float4*)(xr + phase * 1024 + c);
            v4h h = { (_Float16)v.x, (_Float16)v.y, (_Float16)v.z, (_Float16)v.w };
            *(v4h*)&xs[m * RSX + c] = h;
        }
        __syncthreads();
        const _Float16* arow = &xs[l16 * RSX];
        for (int kt2 = 0; kt2 < 32; ++kt2) {
            int kb = kt2 * 32;
            v8h alo = *(const v8h*)(arow + kb + hk * 16);
            v8h ahi = *(const v8h*)(arow + kb + hk * 16 + 8);
            long blk = ((long)t * 8 + wave) * 64 + phase * 32 + kt2;
            const _Float16* bp = WdP + (blk * 32 + lane) * 16;
            v8h blo = *(const v8h*)bp;
            v8h bhi = *(const v8h*)(bp + 8);
            v16h A = CAT16(alo, ahi);
            v16h B = CAT16(blo, bhi);
            acc = __builtin_amdgcn_wmma_f32_16x16x32_f16(
                false, A, false, B, (short)0, acc, false, false);
        }
    }
    // SiLU, store h (f16) in permuted order
    long hbase = (long)(sstart + mBase) * HID + wave * 16 + l16;
    for (int r = 0; r < 8; ++r) {
        int M = r + hk * 8;
        if (mBase + M < cnt) {
            float v = acc[r];
            float s = v / (1.0f + __expf(-v));
            Hp[hbase + (long)M * HID] = (_Float16)s;
        }
    }
}

// ---------------- up projection + residual ----------------
// grid: (256 M-tiles, 8 tasks), block: 256 (8 waves; wave w owns nTiles w, w+8, ...)
#define RSH 136  // LDS row stride in halves (128 + 8 pad)

__global__ __launch_bounds__(256) void k_up(
    const float* __restrict__ x, const _Float16* __restrict__ WuP,
    const float* __restrict__ bu, const _Float16* __restrict__ Hp,
    const int* __restrict__ perm, const int* __restrict__ counts,
    const int* __restrict__ offs, float* __restrict__ out)
{
    __shared__ _Float16 hs[16 * RSH];
    __shared__ int rows[16];
    int t = blockIdx.y;
    int cnt = counts[t];
    int mBase = blockIdx.x * 16;
    if (mBase >= cnt) return;
    int sstart = offs[t];

    int tid = threadIdx.x;
    int lane = tid & 31;
    int wave = tid >> 5;
    int l16 = lane & 15;
    int hk  = lane >> 4;

    {   // stage 16x128 h tile (8 halves per thread)
        int m = tid >> 4;
        int ch = (tid & 15) * 8;
        long ps = (long)(sstart + imin(mBase + m, cnt - 1)) * HID + ch;
        *(v8h*)&hs[m * RSH + ch] = *(const v8h*)&Hp[ps];
        if (tid < 16) rows[tid] = perm[sstart + imin(mBase + tid, cnt - 1)];
    }
    __syncthreads();

    // preload all 4 A fragments (K = 128)
    v16h A[4];
    const _Float16* arow = &hs[l16 * RSH];
    for (int kt = 0; kt < 4; ++kt) {
        v8h lo = *(const v8h*)(arow + kt * 32 + hk * 16);
        v8h hi = *(const v8h*)(arow + kt * 32 + hk * 16 + 8);
        A[kt] = CAT16(lo, hi);
    }

    for (int j = 0; j < 16; ++j) {
        int nT = wave + j * 8;            // 0..127
        int ncol = nT * 16 + l16;
        float bias = bu[t * SIZE + ncol];
        v8f acc = { bias, bias, bias, bias, bias, bias, bias, bias };
        for (int kt = 0; kt < 4; ++kt) {
            long blk = ((long)t * 128 + nT) * 4 + kt;
            const _Float16* bp = WuP + (blk * 32 + lane) * 16;
            v8h blo = *(const v8h*)bp;
            v8h bhi = *(const v8h*)(bp + 8);
            v16h B = CAT16(blo, bhi);
            acc = __builtin_amdgcn_wmma_f32_16x16x32_f16(
                false, A[kt], false, B, (short)0, acc, false, false);
        }
        for (int r = 0; r < 8; ++r) {
            int M = r + hk * 8;
            if (mBase + M < cnt) {
                long o = (long)rows[M] * SIZE + ncol;
                out[o] = x[o] + acc[r];
            }
        }
    }
}

// ---------------- host launch ----------------

extern "C" void kernel_launch(void* const* d_in, const int* in_sizes, int n_in,
                              void* d_out, int out_size, void* d_ws, size_t ws_size,
                              hipStream_t stream) {
    const float* x  = (const float*)d_in[0];
    const float* Wd = (const float*)d_in[1];
    const float* bd = (const float*)d_in[2];
    const float* Wu = (const float*)d_in[3];
    const float* bu = (const float*)d_in[4];
    const int* task_id = (const int*)d_in[5];
    float* out = (float*)d_out;

    // workspace layout (256B aligned bump allocator)
    size_t off = 0;
    auto bump = [&](size_t bytes) -> void* {
        void* p = (char*)d_ws + off;
        off = (off + bytes + 255) & ~(size_t)255;
        return p;
    };
    int* ctrs = (int*)bump(16 * sizeof(int));     // [0..7]=counts, [8..15]=cursors
    int* counts  = ctrs;
    int* cursors = ctrs + 8;
    int* offs = (int*)bump(8 * sizeof(int));
    int* perm = (int*)bump(BATCH * sizeof(int));
    _Float16* WdP = (_Float16*)bump((size_t)NTASK * SIZE * HID * sizeof(_Float16));
    _Float16* WuP = (_Float16*)bump((size_t)NTASK * HID * SIZE * sizeof(_Float16));
    _Float16* Hp  = (_Float16*)bump((size_t)BATCH * HID * sizeof(_Float16));
    (void)ws_size; (void)in_sizes; (void)n_in; (void)out_size;

    // 1) zero counters; residual copy out = x
    k_zero<<<1, 64, 0, stream>>>(ctrs);
    int n4 = BATCH * SIZE / 4;
    k_copy<<<(n4 + 255) / 256, 256, 0, stream>>>((const float4*)x, (float4*)out, n4);

    // 2) bucket rows by task
    k_count<<<BATCH / 256, 256, 0, stream>>>(task_id, counts);
    k_scan<<<1, 1, 0, stream>>>(counts, offs);
    k_scatter<<<BATCH / 256, 256, 0, stream>>>(task_id, offs, cursors, perm);

    // 3) pack weights into f16 WMMA-B layout
    long wdElems = (long)NTASK * SIZE * HID;   // 2,097,152
    k_pack<<<(int)((wdElems + 255) / 256), 256, 0, stream>>>(Wd, WdP, SIZE, HID);
    long wuElems = (long)NTASK * HID * SIZE;
    k_pack<<<(int)((wuElems + 255) / 256), 256, 0, stream>>>(Wu, WuP, HID, SIZE);

    // 4) grouped GEMMs
    dim3 gg(BATCH / 16, NTASK);
    k_down<<<gg, 256, 0, stream>>>(x, WdP, bd, perm, counts, offs, Hp);
    k_up<<<gg, 256, 0, stream>>>(x, WuP, bu, Hp, perm, counts, offs, out);
}
